// AttModel_575525617777
// MI455X (gfx1250) — matrix operand order repaired and checked by reading, not compiled, not run on hardware
//
#include <hip/hip_runtime.h>
#include <math.h>
#include <stdint.h>

typedef _Float16 half_t;
typedef __attribute__((ext_vector_type(16))) _Float16 v16h;
typedef __attribute__((ext_vector_type(8)))  _Float16 v8h;
typedef __attribute__((ext_vector_type(8)))  float    v8f;

#define VOCAB 50000
#define NCLS  6
#define HDIM  512
#define EDIM  300
#define EPAD  320
#define BB    256
#define SS    512
#define H3    1536

__device__ __forceinline__ v16h cat16(v8h lo, v8h hi) {
  return __builtin_shufflevector(lo, hi, 0,1,2,3,4,5,6,7,8,9,10,11,12,13,14,15);
}

// A fragment (16x32 f16): row-major src, leading dim ld (elements).
// k order used here (same as loadB): lane l: row = l%16; lanes<16 hold K=kc..kc+15, lanes>=16 hold K=kc+16..kc+31.
__device__ __forceinline__ v16h loadA(const half_t* __restrict__ src, int ld, int kc, int lane) {
  int row = lane & 15, hf = lane >> 4;
  const v8h* p0 = (const v8h*)(src + (size_t)row * ld + kc + hf * 16);
  const v8h* p1 = (const v8h*)(src + (size_t)row * ld + kc + hf * 16 + 8);
  return cat16(*p0, *p1);
}

// B fragment (32x16 f16): element(k,n) = src[n*ld + k] (column panels contiguous in K).
// k order used here (same as loadA): lane l: col = l%16; lanes<16 hold K=kc..kc+15, lanes>=16 hold K=kc+16..kc+31.
__device__ __forceinline__ v16h loadB(const half_t* __restrict__ src, int ld, int kc, int lane) {
  int col = lane & 15, kb = (lane >> 4) * 16;
  const v8h* p0 = (const v8h*)(src + (size_t)col * ld + kc + kb);
  const v8h* p1 = (const v8h*)(src + (size_t)col * ld + kc + kb + 8);
  return cat16(*p0, *p1);
}

// ---------- prep kernels ----------

__global__ void k_prep_wih(const float* __restrict__ Wih, half_t* __restrict__ WihF) {
  int idx = blockIdx.x * blockDim.x + threadIdx.x;
  if (idx >= H3 * EPAD) return;
  int k = idx % EPAD, n = idx / EPAD;
  WihF[idx] = (k < EDIM) ? (half_t)Wih[(size_t)n * EDIM + k] : (half_t)0.f;
}

__global__ void k_prep_whh(const float* __restrict__ Whh, half_t* __restrict__ WhhF) {
  int idx = blockIdx.x * blockDim.x + threadIdx.x;
  if (idx >= H3 * HDIM) return;
  WhhF[idx] = (half_t)Whh[idx];
}

// ta[c][h] = tanh( tanh(class_emb[c]) @ attend )[h], rows 6..15 zero-padded.
__global__ void k_attnvec(const float* __restrict__ clsW, const float* __restrict__ attend,
                          half_t* __restrict__ ta) {
  int h = threadIdx.x, c = blockIdx.x;
  if (c < NCLS) {
    float acc = 0.f;
    for (int k = 0; k < HDIM; ++k)
      acc += tanhf(clsW[c * HDIM + k]) * attend[(size_t)k * HDIM + h];
    ta[c * HDIM + h] = (half_t)tanhf(acc);
  } else {
    ta[c * HDIM + h] = (half_t)0.f;
  }
}

// emb[s][b][e] = tanh(embed_W[seq[b][s]][e]) (f16, e>=300 zero); mask[s*B+b] = seq>0.
__global__ void k_embed(const int* __restrict__ seq, const float* __restrict__ embW,
                        half_t* __restrict__ emb, float* __restrict__ mask) {
  int idx = blockIdx.x * blockDim.x + threadIdx.x;
  if (idx >= SS * BB * EPAD) return;
  int e = idx % EPAD;
  int r = idx / EPAD;           // r = s*256 + b
  int b = r & 255, s = r >> 8;
  int tok = seq[b * SS + s];
  half_t v = (half_t)0.f;
  if (e < EDIM) v = (half_t)tanhf(embW[(size_t)tok * EDIM + e]);
  emb[(size_t)r * EPAD + e] = v;
  if (e == 0) mask[r] = (tok > 0) ? 1.f : 0.f;
}

// ---------- gi = emb @ W_ih^T + b_ih : WMMA GEMM [131072 x 1536, K=320] ----------
// 64x64 macro-tile per wave (4x4 WMMA tiles): 4x fewer fragment fetches per FLOP.
#define GI_MT 4
#define GI_NT 4
__global__ void __launch_bounds__(256) k_gi(const half_t* __restrict__ emb,
                                            const half_t* __restrict__ WihF,
                                            const float* __restrict__ bih,
                                            half_t* __restrict__ gi) {
  int wid = blockIdx.x * 8 + (threadIdx.x >> 5);   // 49152 waves
  int lane = threadIdx.x & 31;
  const int NTG = 96 / GI_NT;                       // 24 col groups
  int rg = wid / NTG, ng = wid - rg * NTG;
  int r0 = rg * (16 * GI_MT), n0 = ng * (16 * GI_NT);

  v8f acc[GI_MT][GI_NT] = {};
  for (int kc = 0; kc < EPAD; kc += 32) {
    v16h a[GI_MT];
#pragma unroll
    for (int ia = 0; ia < GI_MT; ++ia)
      a[ia] = loadA(emb + (size_t)(r0 + ia * 16) * EPAD, EPAD, kc, lane);
#pragma unroll
    for (int jb = 0; jb < GI_NT; ++jb) {
      v16h b = loadB(WihF + (size_t)(n0 + jb * 16) * EPAD, EPAD, kc, lane);
#pragma unroll
      for (int ia = 0; ia < GI_MT; ++ia)
        acc[ia][jb] = __builtin_amdgcn_wmma_f32_16x16x32_f16(false, a[ia], false, b, (short)0,
                                                             acc[ia][jb], false, false);
    }
  }
  int col = lane & 15, mrow = (lane >> 4) * 8;
#pragma unroll
  for (int jb = 0; jb < GI_NT; ++jb) {
    float bi = bih[n0 + jb * 16 + col];
#pragma unroll
    for (int ia = 0; ia < GI_MT; ++ia)
#pragma unroll
      for (int q = 0; q < 8; ++q)
        gi[(size_t)(r0 + ia * 16 + q + mrow) * H3 + n0 + jb * 16 + col] =
            (half_t)(acc[ia][jb][q] + bi);
  }
}

// ---------- persistent GRU: 512 steps, LDS-resident W_hh panels, global barrier ----------
__global__ void __launch_bounds__(128) k_gru(const half_t* __restrict__ Whh,  // [1536][512] f16
                                             const half_t* __restrict__ gi,   // [S*B][1536] f16
                                             const float* __restrict__ bhh,   // [1536]
                                             half_t* __restrict__ hping,      // [B][512] zeroed
                                             half_t* __restrict__ hpong,
                                             half_t* __restrict__ hs,         // [S*B][512]
                                             int* __restrict__ bar) {
  __shared__ __align__(16) half_t ldsB[3][16][HDIM + 8];
  const int hj = blockIdx.x & 31;
  const int bigrp = blockIdx.x >> 5;
  const int wave = threadIdx.x >> 5;
  const int lane = threadIdx.x & 31;
  const int bi = bigrp * 4 + wave;

  // Seed store: makes ldsB visibly written (prevents undef-folding of its loads);
  // overwritten by the async DMA below.
  if (threadIdx.x == 0) ldsB[0][0][0] = (half_t)0.f;

  // Stage the 48 W_hh column panels (3 gates x 16 cols, 1024B rows) straight into LDS
  // with CDNA5 async global->LDS copies (ASYNCcnt-tracked, no VGPR bounce).
  // LDS offset = low 32 bits of the generic address (ISA 10.2);
  // the ptr->int conversion also marks ldsB as captured by the asm's memory clobber.
  {
    unsigned ldsBase = (unsigned)(uintptr_t)&ldsB[0][0][0];
    const int LDS_STRIDE_B = (HDIM + 8) * 2;  // padded row stride in bytes
    for (int i = threadIdx.x; i < 48 * 64; i += blockDim.x) {  // 64 x 16B chunks per row
      int r = i >> 6;
      int cb = (i & 63) << 4;                  // byte offset within row
      int g = r >> 4, cc = r & 15;
      int n = g * HDIM + hj * 16 + cc;
      unsigned ldsoff = ldsBase + (unsigned)(r * LDS_STRIDE_B + cb);
      unsigned long long ga =
          (unsigned long long)(const void*)(Whh + (size_t)n * HDIM) + (unsigned)cb;
      asm volatile("global_load_async_to_lds_b128 %0, %1, off"
                   :: "v"(ldsoff), "v"(ga) : "memory");
    }
    asm volatile("s_wait_asynccnt 0x0" ::: "memory");
  }
  __syncthreads();

  const int col = lane & 15;
  const int kb = (lane >> 4) * 16;
  const int hcol = hj * 16 + col;
  const int mrow = (lane >> 4) * 8;
  const int b0 = bi * 16;
  const float bhr = bhh[hcol], bhz = bhh[hcol + 512], bhn = bhh[hcol + 1024];

  for (int s = 0; s < SS; ++s) {
    const half_t* hin = (s & 1) ? hpong : hping;
    half_t* hout = (s & 1) ? hping : hpong;
    const half_t* arow = hin + (size_t)b0 * HDIM;
    v8f cr = {}, cz = {}, cn = {};
    for (int kc = 0; kc < HDIM; kc += 32) {
      v16h a = loadA(arow, HDIM, kc, lane);
      v16h br = cat16(*(const v8h*)&ldsB[0][col][kc + kb], *(const v8h*)&ldsB[0][col][kc + kb + 8]);
      v16h bz = cat16(*(const v8h*)&ldsB[1][col][kc + kb], *(const v8h*)&ldsB[1][col][kc + kb + 8]);
      v16h bn = cat16(*(const v8h*)&ldsB[2][col][kc + kb], *(const v8h*)&ldsB[2][col][kc + kb + 8]);
      cr = __builtin_amdgcn_wmma_f32_16x16x32_f16(false, a, false, br, (short)0, cr, false, false);
      cz = __builtin_amdgcn_wmma_f32_16x16x32_f16(false, a, false, bz, (short)0, cz, false, false);
      cn = __builtin_amdgcn_wmma_f32_16x16x32_f16(false, a, false, bn, (short)0, cn, false, false);
    }
#pragma unroll
    for (int q = 0; q < 8; ++q) {
      int b = b0 + q + mrow;
      size_t gidx = ((size_t)s * BB + b) * H3;
      float gir = (float)gi[gidx + hcol];
      float giz = (float)gi[gidx + hcol + 512];
      float gin = (float)gi[gidx + hcol + 1024];
      float hp = (float)hin[(size_t)b * HDIM + hcol];
      float r = 1.f / (1.f + __expf(-(gir + cr[q] + bhr)));
      float z = 1.f / (1.f + __expf(-(giz + cz[q] + bhz)));
      float n = tanhf(gin + r * (cn[q] + bhn));
      float hnew = (1.f - z) * n + z * hp;
      hout[(size_t)b * HDIM + hcol] = (half_t)hnew;
      hs[((size_t)s * BB + b) * HDIM + hcol] = (half_t)hnew;
      // hide next step's gi fetch latency on the serial critical path
      if (s + 1 < SS) __builtin_prefetch(gi + gidx + (size_t)BB * H3 + hcol, 0, 1);
    }
    // device-wide sense-reversal barrier between steps
    __threadfence();
    __syncthreads();
    if (threadIdx.x == 0) {
      int arrived = __hip_atomic_fetch_add(&bar[0], 1, __ATOMIC_ACQ_REL, __HIP_MEMORY_SCOPE_AGENT);
      if (arrived == (int)gridDim.x - 1) {
        __hip_atomic_store(&bar[0], 0, __ATOMIC_RELAXED, __HIP_MEMORY_SCOPE_AGENT);
        __hip_atomic_fetch_add(&bar[1], 1, __ATOMIC_RELEASE, __HIP_MEMORY_SCOPE_AGENT);
      } else {
        while (__hip_atomic_load(&bar[1], __ATOMIC_ACQUIRE, __HIP_MEMORY_SCOPE_AGENT) <= s)
          __builtin_amdgcn_s_sleep(2);
      }
    }
    __syncthreads();
  }
}

// ---------- scores: ts[c][b][s] = tanh(mask * (ta[c] . hs[s,b])) via WMMA ----------
__global__ void __launch_bounds__(256) k_scores(const half_t* __restrict__ ta,   // [16][512]
                                                const half_t* __restrict__ hs,   // [S*B][512]
                                                const float* __restrict__ mask,  // [S*B]
                                                half_t* __restrict__ ts) {       // [16][B][S]
  int wid = blockIdx.x * 8 + (threadIdx.x >> 5);
  int lane = threadIdx.x & 31;
  int n0 = wid * 16;
  v8f acc = {};
  for (int kc = 0; kc < HDIM; kc += 32) {
    v16h a = loadA(ta, HDIM, kc, lane);
    v16h b = loadB(hs + (size_t)n0 * HDIM, HDIM, kc, lane);
    acc = __builtin_amdgcn_wmma_f32_16x16x32_f16(false, a, false, b, (short)0, acc, false, false);
  }
  int col = lane & 15, mrow = (lane >> 4) * 8;
  int n = n0 + col;
  int sIdx = n >> 8, bIdx = n & 255;
  float m = mask[n];
#pragma unroll
  for (int q = 0; q < 8; ++q) {
    int c = q + mrow;
    ts[(size_t)c * (BB * SS) + (size_t)bIdx * SS + sIdx] = (half_t)tanhf(acc[q] * m);
  }
}

// ---------- ctx[c][b][h] = sum_s ts[c][b][s] * hs[s][b][h] ----------
__global__ void k_ctx(const half_t* __restrict__ ts, const half_t* __restrict__ hs,
                      float* __restrict__ ctx) {
  int idx = blockIdx.x * blockDim.x + threadIdx.x;
  if (idx >= NCLS * BB * HDIM) return;
  int h = idx & 511, b = (idx >> 9) & 255, c = idx >> 17;
  float acc = 0.f;
  for (int s = 0; s < SS; ++s)
    acc += (float)ts[(size_t)c * (BB * SS) + b * SS + s] * (float)hs[((size_t)s * BB + b) * HDIM + h];
  ctx[idx] = acc;
}

// ---------- logits + log_softmax ----------
__global__ void k_logits(const float* __restrict__ ctx, const float* __restrict__ Wcls,
                         const float* __restrict__ bcls, float* __restrict__ out) {
  int t = blockIdx.x * blockDim.x + threadIdx.x;
  if (t >= NCLS * BB) return;
  const float* cx = ctx + (size_t)t * HDIM;
  float l0 = bcls[0], l1 = bcls[1];
  for (int h = 0; h < HDIM; ++h) {
    float v = tanhf(cx[h]);
    l0 += v * Wcls[h];
    l1 += v * Wcls[HDIM + h];
  }
  float m = fmaxf(l0, l1);
  float lse = m + logf(expf(l0 - m) + expf(l1 - m));
  out[t * 2 + 0] = l0 - lse;
  out[t * 2 + 1] = l1 - lse;
}

extern "C" void kernel_launch(void* const* d_in, const int* in_sizes, int n_in,
                              void* d_out, int out_size, void* d_ws, size_t ws_size,
                              hipStream_t stream) {
  (void)in_sizes; (void)n_in; (void)out_size; (void)ws_size;
  const int*   seq    = (const int*)d_in[0];
  const float* embW   = (const float*)d_in[2];
  const float* clsW   = (const float*)d_in[3];
  const float* Wih    = (const float*)d_in[4];
  const float* Whh    = (const float*)d_in[5];
  const float* bih    = (const float*)d_in[6];
  const float* bhh    = (const float*)d_in[7];
  const float* attend = (const float*)d_in[8];
  const float* Wcls   = (const float*)d_in[9];
  const float* bcls   = (const float*)d_in[10];

  char* p = (char*)d_ws;
  auto alloc = [&](size_t bytes) { char* r = p; p += (bytes + 255) & ~(size_t)255; return r; };
  half_t* emb   = (half_t*)alloc((size_t)SS * BB * EPAD * 2);   // 84 MB
  half_t* WihF  = (half_t*)alloc((size_t)H3 * EPAD * 2);
  half_t* WhhF  = (half_t*)alloc((size_t)H3 * HDIM * 2);
  half_t* gi    = (half_t*)alloc((size_t)SS * BB * H3 * 2);     // 403 MB
  half_t* hs    = (half_t*)alloc((size_t)SS * BB * HDIM * 2);   // 134 MB
  half_t* hping = (half_t*)alloc((size_t)BB * HDIM * 2);
  half_t* hpong = (half_t*)alloc((size_t)BB * HDIM * 2);
  half_t* ta    = (half_t*)alloc((size_t)16 * HDIM * 2);
  float*  mask  = (float*)alloc((size_t)SS * BB * 4);
  half_t* ts    = (half_t*)alloc((size_t)16 * BB * SS * 2);
  float*  ctx   = (float*)alloc((size_t)NCLS * BB * HDIM * 4);
  int*    bar   = (int*)alloc(256);

  hipMemsetAsync(hping, 0, (size_t)BB * HDIM * 2, stream);
  hipMemsetAsync(bar, 0, 256, stream);

  k_prep_wih<<<(H3 * EPAD + 255) / 256, 256, 0, stream>>>(Wih, WihF);
  k_prep_whh<<<(H3 * HDIM + 255) / 256, 256, 0, stream>>>(Whh, WhhF);
  k_attnvec<<<16, 512, 0, stream>>>(clsW, attend, ta);
  k_embed<<<(SS * BB * EPAD + 255) / 256, 256, 0, stream>>>(seq, embW, emb, mask);
  k_gi<<<(2048 * 24) / 8, 256, 0, stream>>>(emb, WihF, bih, gi);
  k_gru<<<128, 128, 0, stream>>>(WhhF, gi, bhh, hping, hpong, hs, bar);
  k_scores<<<8192 / 8, 256, 0, stream>>>(ta, hs, mask, ts);
  k_ctx<<<(NCLS * BB * HDIM + 255) / 256, 256, 0, stream>>>(ts, hs, ctx);
  k_logits<<<(NCLS * BB + 255) / 256, 256, 0, stream>>>(ctx, Wcls, bcls, (float*)d_out);
}
